// MyLoss_17179869184310
// MI455X (gfx1250) — matrix operand order repaired and checked by reading, not compiled, not run on hardware
//
#include <hip/hip_runtime.h>

typedef _Float16 v4h  __attribute__((ext_vector_type(4)));
typedef _Float16 v8h  __attribute__((ext_vector_type(8)));
typedef _Float16 v16h __attribute__((ext_vector_type(16)));
typedef float    v8f  __attribute__((ext_vector_type(8)));
typedef unsigned int u32;
typedef u32 u32x4 __attribute__((ext_vector_type(4)));
typedef u32 u32x8 __attribute__((ext_vector_type(8)));

namespace {
constexpr int kRows  = 8192;   // 2b
constexpr int kCols  = 1024;   // feature dim
constexpr int kHalfB = 4096;   // b
// TDM staging geometry for the fused kernel
constexpr int kGrpCols   = 128;                 // columns staged per phase
constexpr int kChunkK    = 128;                 // K elements staged per phase
constexpr int kColStride = kChunkK + 8;         // 136 f16 = 272 B (16 B pad/row)
}  // namespace

// ---------------------------------------------------------------------------
// Kernel 1: row L2-normalize, emit f16 copy of y + reciprocal norms.
// ---------------------------------------------------------------------------
__global__ void __launch_bounds__(256) normalize_kernel(
    const float* __restrict__ x, _Float16* __restrict__ yh,
    float* __restrict__ rinv) {
  const int row = blockIdx.x;
  const int tid = threadIdx.x;
  const float4 a = reinterpret_cast<const float4*>(x + (size_t)row * kCols)[tid];
  float s = a.x * a.x + a.y * a.y + a.z * a.z + a.w * a.w;
#pragma unroll
  for (int m = 16; m >= 1; m >>= 1) s += __shfl_xor(s, m, 32);
  __shared__ float red[8];
  if ((tid & 31) == 0) red[tid >> 5] = s;
  __syncthreads();
  float total = 0.0f;
#pragma unroll
  for (int i = 0; i < 8; ++i) total += red[i];
  const float r = rsqrtf(fmaxf(total, 1e-12f));
  if (tid == 0) rinv[row] = r;
  v4h o;
  o.x = (_Float16)(a.x * r);
  o.y = (_Float16)(a.y * r);
  o.z = (_Float16)(a.z * r);
  o.w = (_Float16)(a.w * r);
  reinterpret_cast<v4h*>(yh + (size_t)row * kCols)[tid] = o;
}

// ---------------------------------------------------------------------------
// Kernel 2: zero the per-row expsum accumulators and the loss slot.
// ---------------------------------------------------------------------------
__global__ void init_kernel(float* __restrict__ ws_sum,
                            float* __restrict__ loss_slot) {
  const int i = blockIdx.x * blockDim.x + threadIdx.x;
  if (i < kRows) ws_sum[i] = 0.0f;
  if (i == 0) *loss_slot = 0.0f;
}

// ---------------------------------------------------------------------------
// Kernel 3: fp32 label dots: d[i] = y_i . y_{i+b}; cos output.
// ---------------------------------------------------------------------------
__global__ void __launch_bounds__(256) cos_kernel(
    const float* __restrict__ x, const float* __restrict__ rinv,
    float* __restrict__ cos_out, float* __restrict__ dlab) {
  const int i = blockIdx.x;
  const int tid = threadIdx.x;
  const float4 va = reinterpret_cast<const float4*>(x + (size_t)i * kCols)[tid];
  const float4 vb =
      reinterpret_cast<const float4*>(x + (size_t)(i + kHalfB) * kCols)[tid];
  float s = va.x * vb.x + va.y * vb.y + va.z * vb.z + va.w * vb.w;
#pragma unroll
  for (int m = 16; m >= 1; m >>= 1) s += __shfl_xor(s, m, 32);
  __shared__ float red[8];
  if ((tid & 31) == 0) red[tid >> 5] = s;
  __syncthreads();
  if (tid == 0) {
    float total = 0.0f;
#pragma unroll
    for (int w = 0; w < 8; ++w) total += red[w];
    const float d = total * rinv[i] * rinv[i + kHalfB];
    cos_out[i] = (1.0f + d) * 0.5f;
    dlab[i] = d;
    dlab[i + kHalfB] = d;
  }
}

// ---------------------------------------------------------------------------
// Kernel 4: fused  sim = y.y^T  (f16 WMMA, f32 acc)  +  row logsumexp.
//
// B panels (128 cols x 128 K) staged into LDS by the Tensor Data Mover,
// double-buffered on TENSORcnt; one barrier per phase (TDM store for a
// buffer is issued only after the barrier proving all waves left it).
// Fragment loads are software-pipelined through double register buffers so
// each WMMA waits on loads issued a full step earlier, not just-in-time.
// ---------------------------------------------------------------------------
__global__ void __launch_bounds__(256) simclr_lse_kernel(
    const _Float16* __restrict__ yh, float* __restrict__ ws_sum) {
  __shared__ __align__(16) _Float16 Bs[2][kGrpCols * kColStride];  // 68 KB

  const int lane = threadIdx.x & 31;
  const int wave = threadIdx.x >> 5;
  const int half = lane >> 4;
  const int l15  = lane & 15;
  const int rowbase  = blockIdx.x * 128 + wave * 16;
  const int colstart = blockIdx.y * 2048;

  // A fragment base: lane holds row (rowbase+l15); per 32-K step it needs
  // K = kpos + {half*8+0..7, 16+half*8+0..7}  -> two 16B chunks.
  const _Float16* arow = yh + (size_t)(rowbase + l15) * kCols + half * 8;

  float sumexp[8] = {0.f, 0.f, 0.f, 0.f, 0.f, 0.f, 0.f, 0.f};

  // Issue one TDM load: chunk idx covers cols [colstart+(idx>>3)*128, +128),
  // K = [(idx&7)*128, +128), into LDS buffer (idx&1).
  auto stage = [&](int idx) {
    const int gg  = idx >> 3;
    const int cc  = idx & 7;
    const int buf = idx & 1;
    const _Float16* gp =
        yh + ((size_t)(colstart + gg * kGrpCols) * kCols + cc * kChunkK);
    const unsigned long long ga = reinterpret_cast<unsigned long long>(gp);
    u32x4 g0;
    g0[0] = 1u;  // count=1, no gather, user descriptor
    g0[1] = (u32)reinterpret_cast<unsigned long long>(&Bs[buf][0]);  // LDS addr
    g0[2] = (u32)ga;                                        // global_addr[31:0]
    g0[3] = ((u32)(ga >> 32) & 0x01FFFFFFu) | 0x80000000u;  // addr[56:32]|type=2
    u32x8 g1;
    // data_size=2B (1<<16) | pad_enable (1<<20) | pad_interval=64dw (5<<22)
    // | pad_amount=4dw (3<<25)
    g1[0] = 0x07510000u;
    g1[1] = (u32)kCols << 16;              // tensor_dim0 = 1024 (bits 79:48)
    g1[2] = ((u32)kRows & 0xFFFFu) << 16;  // tensor_dim1 = 8192 (bits 111:80)
    g1[3] = (u32)kChunkK << 16;            // tile_dim0 = 128  (bits 127:112)
    g1[4] = (u32)kGrpCols;                 // tile_dim1 = 128  (bits 143:128)
    g1[5] = (u32)kCols;                    // tensor_dim0_stride = 1024
    g1[6] = 0u;
    g1[7] = 0u;
    asm volatile("tensor_load_to_lds %0, %1" ::"s"(g0), "s"(g1) : "memory");
  };

  if (wave == 0) stage(0);

  union Frag { v16h v; v8h h[2]; };

  for (int g = 0; g < 16; ++g) {  // 16 groups of 128 columns per split
    v8f c[8] = {};
    for (int cc = 0; cc < 8; ++cc) {  // 8 K-chunks of 128
      const int idx = g * 8 + cc;
      if (wave == 0) __builtin_amdgcn_s_wait_tensorcnt(0);  // chunk idx ready
      __syncthreads();
      // All waves are past their use of buffer ((idx+1)&1) (phase idx-1),
      // so it is now safe to start overwriting it.
      if (wave == 0 && idx + 1 < 128) stage(idx + 1);

      const _Float16* bbase = &Bs[idx & 1][0];
      Frag A[2], B[2];
      auto loadA = [&](Frag& dst, int kb) {
        const _Float16* ak = arow + (cc * kChunkK + kb * 32);
        dst.h[0] = *reinterpret_cast<const v8h*>(ak);
        dst.h[1] = *reinterpret_cast<const v8h*>(ak + 16);
      };
      auto loadB = [&](Frag& dst, int j) {  // j = kb*8 + t
        const _Float16* bp = bbase + ((j & 7) * 16 + l15) * kColStride +
                             (j >> 3) * 32 + half * 8;
        dst.h[0] = *reinterpret_cast<const v8h*>(bp);
        dst.h[1] = *reinterpret_cast<const v8h*>(bp + 16);
      };
      loadA(A[0], 0);
      loadB(B[0], 0);
#pragma unroll
      for (int j = 0; j < 32; ++j) {  // 4 kb-groups x 8 column tiles
        const int kb = j >> 3;
        const int t  = j & 7;
        if (t == 0 && kb < 3) loadA(A[(kb + 1) & 1], kb + 1);  // 8-WMMA ahead
        if (j + 1 < 32) loadB(B[(j + 1) & 1], j + 1);          // 1-WMMA ahead
        c[t] = __builtin_amdgcn_wmma_f32_16x16x32_f16(
            false, A[kb & 1].v, false, B[j & 1].v, (short)0, c[t], false,
            false);
      }
    }
    // Accumulate exp(10*sim - 10) per C slot; mask the diagonal exactly.
    const int colbase = colstart + g * 128;
#pragma unroll
    for (int t = 0; t < 8; ++t) {
      const int col = colbase + t * 16 + l15;
#pragma unroll
      for (int r = 0; r < 8; ++r) {
        const int row = rowbase + r + 8 * half;
        const float e = __expf(10.0f * c[t][r] - 10.0f);
        sumexp[r] += (row == col) ? 0.0f : e;
      }
    }
  }
  // Reduce the 16 N-slots of each row (lanes within each 16-lane half).
#pragma unroll
  for (int r = 0; r < 8; ++r) {
    float s = sumexp[r];
    s += __shfl_xor(s, 1, 32);
    s += __shfl_xor(s, 2, 32);
    s += __shfl_xor(s, 4, 32);
    s += __shfl_xor(s, 8, 32);
    sumexp[r] = s;
  }
  if (l15 == 0) {
#pragma unroll
    for (int r = 0; r < 8; ++r)
      atomicAdd(&ws_sum[rowbase + r + 8 * half], sumexp[r]);
  }
}

// ---------------------------------------------------------------------------
// Kernel 5: loss = mean_i (10 + log(sum_i) - 10*dlab[i]).
// ---------------------------------------------------------------------------
__global__ void __launch_bounds__(256) loss_kernel(
    const float* __restrict__ ws_sum, const float* __restrict__ dlab,
    float* __restrict__ loss_slot) {
  const int i = blockIdx.x * 256 + threadIdx.x;
  float li = 10.0f + __logf(ws_sum[i]) - 10.0f * dlab[i];
#pragma unroll
  for (int m = 16; m >= 1; m >>= 1) li += __shfl_xor(li, m, 32);
  __shared__ float red[8];
  if ((threadIdx.x & 31) == 0) red[threadIdx.x >> 5] = li;
  __syncthreads();
  if (threadIdx.x == 0) {
    float total = 0.0f;
#pragma unroll
    for (int w = 0; w < 8; ++w) total += red[w];
    atomicAdd(loss_slot, total * (1.0f / 8192.0f));
  }
}

// ---------------------------------------------------------------------------
extern "C" void kernel_launch(void* const* d_in, const int* in_sizes, int n_in,
                              void* d_out, int out_size, void* d_ws,
                              size_t ws_size, hipStream_t stream) {
  (void)in_sizes; (void)n_in; (void)out_size; (void)ws_size;
  const float* x = (const float*)d_in[0];
  float* out = (float*)d_out;  // [0..4095]=cos, [4096]=loss

  char* ws = (char*)d_ws;
  _Float16* yh   = (_Float16*)ws;                              // 16 MiB
  float*    rinv = (float*)(ws + (size_t)kRows * kCols * 2);   // 32 KiB
  float*    dlab = rinv + kRows;                               // 32 KiB
  float*    wsum = dlab + kRows;                               // 32 KiB

  normalize_kernel<<<kRows, 256, 0, stream>>>(x, yh, rinv);
  init_kernel<<<(kRows + 255) / 256, 256, 0, stream>>>(wsum, out + kHalfB);
  cos_kernel<<<kHalfB, 256, 0, stream>>>(x, rinv, out, dlab);
  dim3 g3(kRows / 128, 4);  // 64 row-blocks x 4 column splits
  simclr_lse_kernel<<<g3, 256, 0, stream>>>(yh, wsum);
  loss_kernel<<<kRows / 256, 256, 0, stream>>>(wsum, dlab, out + kHalfB);
}
